// cls_model_14508399526633
// MI455X (gfx1250) — matrix operand order repaired and checked by reading, not compiled, not run on hardware
//
#include <hip/hip_runtime.h>
#include <hip/hip_bf16.h>

// ---------------------------------------------------------------------------
// Point-cloud cls model forward for MI455X (gfx1250, wave32, WMMA).
//   K1: conv1(3->64)+relu (VALU->LDS f16), conv2(64->64) via WMMA,
//       bias+relu+8:1 maxpool fused in the accumulator -> h16
//   K2: q/k/v projections via WMMA (+ per-row |h|^2)
//   K3: fused flash-softmax (base 2) + KNN top-10 + gathered weighted-V (WMMA)
//   K4: global max-pool over N + final 64->3 linear
// ---------------------------------------------------------------------------

typedef __attribute__((ext_vector_type(16))) _Float16 v16h;
typedef __attribute__((ext_vector_type(8)))  _Float16 v8h;
typedef __attribute__((ext_vector_type(8)))  float    v8f;

#define BATCH 4
#define NIN   32768
#define NPTS  4096
#define DIM   64
#define KN    10
#define QSCALE (0.125f * 1.44269504f)   // 1/sqrt(64) * log2(e): softmax in base 2

// Raw v_exp_f32 (base-2 exponential) — device-only, one instruction.
__device__ __forceinline__ float fast_exp2(float x) {
  return __builtin_amdgcn_exp2f(x);
}

// ---------------------------------------------------------------------------
// WMMA fragment loaders (16x16x32 f16).
// A (M x K): lane l -> row M = l&15; elem[i]      = K = kofs + ((l>=16)?16:0) + i
// B (K x N): lane l -> col N = l&15; elem[i]      = K = kofs + ((l>=16)?16:0) + i
// ---------------------------------------------------------------------------
__device__ __forceinline__ v16h load_a_frag(const _Float16* __restrict__ row, int lane, int kofs) {
  const int kb = (lane & 16) ? 16 : 0;
  const v8h lo = *(const v8h*)(row + kofs + kb);
  const v8h hi = *(const v8h*)(row + kofs + kb + 8);
  return __builtin_shufflevector(lo, hi, 0,1,2,3,4,5,6,7,8,9,10,11,12,13,14,15);
}
__device__ __forceinline__ v16h load_b_frag(const _Float16* __restrict__ row, int lane, int kofs) {
  const int kb = (lane & 16) ? 16 : 0;
  const v8h lo = *(const v8h*)(row + kofs + kb);
  const v8h hi = *(const v8h*)(row + kofs + kb + 8);
  return __builtin_shufflevector(lo, hi, 0,1,2,3,4,5,6,7,8,9,10,11,12,13,14,15);
}

// ---------------------------------------------------------------------------
// K1: one wave per 128 input points (= 16 pooled points).
//   conv1 (3->64, VALU) -> LDS f16 [128x64]
//   conv2 (64->64) via WMMA; maxpool(8) = in-lane max over the 8 accum rows.
// ---------------------------------------------------------------------------
__global__ void __launch_bounds__(32)
k1_mlp_pool(const float* __restrict__ x,
            const float* __restrict__ w1, const float* __restrict__ b1,
            const float* __restrict__ w2, const float* __restrict__ b2,
            _Float16* __restrict__ h16) {
  __shared__ __align__(16) _Float16 sH1[128 * DIM];   // 16 KB
  __shared__ float sW1[DIM * 3];
  __shared__ float sB1[DIM];
  const int lane = threadIdx.x;
  const int c = blockIdx.x;                            // 128-point chunk
  for (int i = lane; i < DIM * 3; i += 32) sW1[i] = w1[i];
  for (int i = lane; i < DIM; i += 32)     sB1[i] = b1[i];
  __syncthreads();

  // conv1: 4 rows per lane, packed v8h stores into LDS
  #pragma unroll 1
  for (int j = 0; j < 4; ++j) {
    const int row = lane * 4 + j;
    const float* xp = x + ((size_t)c * 128 + row) * 3;
    const float x0 = xp[0], x1 = xp[1], x2 = xp[2];
    #pragma unroll
    for (int ch0 = 0; ch0 < DIM; ch0 += 8) {
      v8h pack;
      #pragma unroll
      for (int i = 0; i < 8; ++i) {
        const int ch = ch0 + i;
        const float a = sB1[ch] + sW1[ch * 3 + 0] * x0 + sW1[ch * 3 + 1] * x1
                                + sW1[ch * 3 + 2] * x2;
        pack[i] = (_Float16)fmaxf(a, 0.0f);
      }
      *(v8h*)&sH1[row * DIM + ch0] = pack;
    }
  }
  __syncthreads();

  // W2 B-fragments (f32 -> f16 on the fly) + bias per column tile
  v16h wb[4][2];
  float bcol[4];
  #pragma unroll
  for (int ct = 0; ct < 4; ++ct) {
    const int col = ct * 16 + (lane & 15);
    bcol[ct] = b2[col];
    const int kb = (lane & 16) ? 16 : 0;
    #pragma unroll
    for (int kf = 0; kf < 2; ++kf) {
      v16h f;
      #pragma unroll
      for (int i = 0; i < 16; ++i) f[i] = (_Float16)w2[col * DIM + kf * 32 + kb + i];
      wb[ct][kf] = f;
    }
  }

  // conv2 WMMA + bias + relu + pool:  max(0, max_r(D[r]+bias)) == pooled relu
  const int prow_base = c * 16 + (lane >> 4);
  #pragma unroll 2
  for (int g = 0; g < 8; ++g) {
    const _Float16* arow = &sH1[(g * 16 + (lane & 15)) * DIM];
    const v16h a0 = load_a_frag(arow, lane, 0);
    const v16h a1 = load_a_frag(arow, lane, 32);
    #pragma unroll
    for (int ct = 0; ct < 4; ++ct) {
      v8f D = {};
      D = __builtin_amdgcn_wmma_f32_16x16x32_f16(false, a0, false, wb[ct][0], (short)0, D, false, false);
      D = __builtin_amdgcn_wmma_f32_16x16x32_f16(false, a1, false, wb[ct][1], (short)0, D, false, false);
      float pm = 0.0f;
      #pragma unroll
      for (int r = 0; r < 8; ++r) pm = fmaxf(pm, D[r] + bcol[ct]);
      h16[(size_t)(prow_base + 2 * g) * DIM + ct * 16 + (lane & 15)] = (_Float16)pm;
    }
  }
}

// ---------------------------------------------------------------------------
// K2: one wave per 16 pooled points. q/k/v via WMMA; per-row |h|^2.
// ---------------------------------------------------------------------------
__global__ void __launch_bounds__(32)
k2_qkv(const _Float16* __restrict__ h16,
       const float* __restrict__ wq, const float* __restrict__ bq,
       const float* __restrict__ wk, const float* __restrict__ bk,
       const float* __restrict__ wv, const float* __restrict__ bv,
       _Float16* __restrict__ q16, _Float16* __restrict__ k16,
       float* __restrict__ v32, float* __restrict__ nrm) {
  const int lane = threadIdx.x;
  const int t = blockIdx.x;                         // 16-row tile (0..1023)
  const int row = t * 16 + (lane & 15);
  const _Float16* hrow = h16 + (size_t)row * DIM;
  const v16h a0 = load_a_frag(hrow, lane, 0);
  const v16h a1 = load_a_frag(hrow, lane, 32);

  // |h|^2: lanes l and l^16 hold disjoint K-halves of row l&15
  float ns = 0.0f;
  #pragma unroll
  for (int i = 0; i < 16; ++i)
    ns += (float)a0[i] * (float)a0[i] + (float)a1[i] * (float)a1[i];
  ns += __shfl_xor(ns, 16, 32);
  if (lane < 16) nrm[row] = ns;

  const int rbase = t * 16 + ((lane >> 4) << 3);
  const int colq  = lane & 15;
  const int kb    = (lane & 16) ? 16 : 0;

  #pragma unroll
  for (int mat = 0; mat < 3; ++mat) {
    const float* W  = (mat == 0) ? wq : (mat == 1) ? wk : wv;
    const float* Bv = (mat == 0) ? bq : (mat == 1) ? bk : bv;
    #pragma unroll
    for (int ct = 0; ct < 4; ++ct) {
      const int col = ct * 16 + colq;
      const float bias = Bv[col];
      v16h w0, w1f;
      #pragma unroll
      for (int i = 0; i < 16; ++i) {
        w0[i]  = (_Float16)W[col * DIM + kb + i];
        w1f[i] = (_Float16)W[col * DIM + 32 + kb + i];
      }
      v8f D = {};
      D = __builtin_amdgcn_wmma_f32_16x16x32_f16(false, a0, false, w0,  (short)0, D, false, false);
      D = __builtin_amdgcn_wmma_f32_16x16x32_f16(false, a1, false, w1f, (short)0, D, false, false);
      #pragma unroll
      for (int r = 0; r < 8; ++r) {
        const float val = D[r] + bias;
        const size_t idx = (size_t)(rbase + r) * DIM + col;
        if (mat == 0)      q16[idx] = (_Float16)(val * QSCALE);
        else if (mat == 1) k16[idx] = (_Float16)val;
        else               v32[idx] = val;
      }
    }
  }
}

// Branchless insertion of (cd,cs,ci) into ascending-sorted top-KN arrays.
__device__ __forceinline__ void topk_insert(float cd, float cs, int ci,
                                            float (&dl)[KN], float (&sl)[KN], int (&il)[KN]) {
  #pragma unroll
  for (int j = 0; j < KN; ++j) {
    const bool sw = cd < dl[j];
    const float nd = sw ? cd : dl[j];
    const float xd = sw ? dl[j] : cd;
    const float ns = sw ? cs : sl[j];
    const float xs = sw ? sl[j] : cs;
    const int   ni = sw ? ci : il[j];
    const int   xi = sw ? il[j] : ci;
    dl[j] = nd; sl[j] = ns; il[j] = ni;
    cd = xd; cs = xs; ci = xi;
  }
}

// ---------------------------------------------------------------------------
// K3: one wave per 16-query tile. Streams over all key tiles:
//   S = K_m . Q_n^T  (scores, pre-scaled by log2e/8) -> base-2 online softmax
//   G = H_m . H_n^T  (Gram) -> rank by  d2' = |h_m|^2 - 2G  (|h_n|^2 constant)
// Per-lane sorted top-10 (index + raw score), lane halves merged via
// shfl_xor(16), then cooperative gather of V rows weighted by softmax@knn.
// ---------------------------------------------------------------------------
__global__ void __launch_bounds__(32)
k3_attn_knn(const _Float16* __restrict__ q16, const _Float16* __restrict__ k16,
            const _Float16* __restrict__ h16, const float* __restrict__ norm,
            const float* __restrict__ v32, float* __restrict__ out32) {
  const int lane  = threadIdx.x;
  const int b     = blockIdx.x >> 8;       // NPTS/16 = 256 tiles per batch
  const int ntile = blockIdx.x & 255;
  const int nq    = ntile * 16 + (lane & 15);
  const size_t base = (size_t)b * NPTS;

  const _Float16* qrow  = q16 + (base + nq) * DIM;
  const _Float16* hnrow = h16 + (base + nq) * DIM;
  const v16h qb0 = load_b_frag(qrow,  lane, 0);
  const v16h qb1 = load_b_frag(qrow,  lane, 32);
  const v16h hb0 = load_b_frag(hnrow, lane, 0);
  const v16h hb1 = load_b_frag(hnrow, lane, 32);

  float mx = -3.0e38f, sm = 0.0f;
  float dl[KN]; float sl[KN]; int il[KN];
  #pragma unroll
  for (int j = 0; j < KN; ++j) { dl[j] = 3.0e38f; sl[j] = 0.0f; il[j] = 0; }

  const int rsub = (lane >> 4) << 3;       // key-row sub-offset: 0 or 8

  #pragma unroll 2
  for (int mt = 0; mt < NPTS / 16; ++mt) {
    const int rowm = mt * 16 + (lane & 15);
    const _Float16* krow = k16 + (base + rowm) * DIM;
    const _Float16* hrow = h16 + (base + rowm) * DIM;
    const v16h ka0 = load_a_frag(krow, lane, 0);
    const v16h ka1 = load_a_frag(krow, lane, 32);
    const v16h ha0 = load_a_frag(hrow, lane, 0);
    const v16h ha1 = load_a_frag(hrow, lane, 32);
    __builtin_prefetch(krow + 16 * DIM, 0, 1);   // next key tile -> global_prefetch_b8
    __builtin_prefetch(hrow + 16 * DIM, 0, 1);

    v8f S = {};
    S = __builtin_amdgcn_wmma_f32_16x16x32_f16(false, ka0, false, qb0, (short)0, S, false, false);
    S = __builtin_amdgcn_wmma_f32_16x16x32_f16(false, ka1, false, qb1, (short)0, S, false, false);
    v8f G = {};
    G = __builtin_amdgcn_wmma_f32_16x16x32_f16(false, ha0, false, hb0, (short)0, G, false, false);
    G = __builtin_amdgcn_wmma_f32_16x16x32_f16(false, ha1, false, hb1, (short)0, G, false, false);

    // --- base-2 online softmax over the 8 scores (branchless tile rescale) --
    float tmax = S[0];
    #pragma unroll
    for (int r = 1; r < 8; ++r) tmax = fmaxf(tmax, S[r]);
    const float nmx = fmaxf(mx, tmax);
    float acc = 0.0f;
    #pragma unroll
    for (int r = 0; r < 8; ++r) acc += fast_exp2(S[r] - nmx);
    sm = sm * fast_exp2(mx - nmx) + acc;
    mx = nmx;

    // --- KNN candidates: rank by norm[m] - 2*G (self excluded) --------------
    #pragma unroll
    for (int r = 0; r < 8; ++r) {
      const int   m  = mt * 16 + rsub + r;
      const float d2 = __builtin_fmaf(-2.0f, G[r], norm[base + m]);
      if (m != nq && d2 < dl[KN - 1]) topk_insert(d2, S[r], m, dl, sl, il);
    }
  }

  // --- merge lane l with lane l^16 (same query, disjoint key rows) ---------
  {
    const float mx2 = __shfl_xor(mx, 16, 32);
    const float sm2 = __shfl_xor(sm, 16, 32);
    const float MX  = fmaxf(mx, mx2);
    sm = sm * fast_exp2(mx - MX) + sm2 * fast_exp2(mx2 - MX);
    mx = MX;
  }
  float pd[KN], ps[KN]; int pi[KN];
  #pragma unroll
  for (int j = 0; j < KN; ++j) {
    pd[j] = __shfl_xor(dl[j], 16, 32);
    ps[j] = __shfl_xor(sl[j], 16, 32);
    pi[j] = __shfl_xor(il[j], 16, 32);
  }
  #pragma unroll
  for (int j = 0; j < KN; ++j) topk_insert(pd[j], ps[j], pi[j], dl, sl, il);

  // --- broadcast final weights / indices, cooperative V gather -------------
  __shared__ float s_w[16 * KN];
  __shared__ int   s_i[16 * KN];
  if (lane < 16) {
    const float inv = 1.0f / sm;
    #pragma unroll
    for (int j = 0; j < KN; ++j) {
      s_w[lane * KN + j] = fast_exp2(sl[j] - mx) * inv;
      s_i[lane * KN + j] = il[j];
    }
  }
  __syncthreads();

  #pragma unroll 1
  for (int q = 0; q < 16; ++q) {
    float a0 = 0.0f, a1 = 0.0f;
    #pragma unroll
    for (int j = 0; j < KN; ++j) {
      const float w = s_w[q * KN + j];
      const float* vr = v32 + (base + (size_t)s_i[q * KN + j]) * DIM;
      a0 += w * vr[lane];
      a1 += w * vr[lane + 32];
    }
    float* orow = out32 + (base + (size_t)(ntile * 16 + q)) * DIM;
    orow[lane]      = a0;
    orow[lane + 32] = a1;
  }
}

// ---------------------------------------------------------------------------
// K4: global max over N + final 64->3 linear. One block per batch.
// ---------------------------------------------------------------------------
__global__ void __launch_bounds__(256)
k4_pool_linear(const float* __restrict__ out32,
               const float* __restrict__ lw, const float* __restrict__ lb,
               float* __restrict__ out) {
  const int b = blockIdx.x;
  const int t = threadIdx.x;
  const int d = t & 63, part = t >> 6;
  const float* src = out32 + ((size_t)b * NPTS + (size_t)part * (NPTS / 4)) * DIM + d;
  float m = -3.0e38f;
  for (int n = 0; n < NPTS / 4; ++n) m = fmaxf(m, src[(size_t)n * DIM]);
  __shared__ float s_m[256];
  s_m[t] = m;
  __syncthreads();
  if (t < 64) s_m[t] = fmaxf(fmaxf(s_m[t], s_m[t + 64]), fmaxf(s_m[t + 128], s_m[t + 192]));
  __syncthreads();
  if (t < 3) {
    float acc = lb[t];
    for (int dd = 0; dd < DIM; ++dd) acc += s_m[dd] * lw[t * DIM + dd];
    out[b * 3 + t] = acc;
  }
}

// ---------------------------------------------------------------------------
extern "C" void kernel_launch(void* const* d_in, const int* in_sizes, int n_in,
                              void* d_out, int out_size, void* d_ws, size_t ws_size,
                              hipStream_t stream) {
  const float* x  = (const float*)d_in[0];
  const float* w1 = (const float*)d_in[1];
  const float* b1 = (const float*)d_in[2];
  const float* w2 = (const float*)d_in[3];
  const float* b2 = (const float*)d_in[4];
  const float* wq = (const float*)d_in[5];
  const float* bq = (const float*)d_in[6];
  const float* wk = (const float*)d_in[7];
  const float* bk = (const float*)d_in[8];
  const float* wv = (const float*)d_in[9];
  const float* bv = (const float*)d_in[10];
  const float* lw = (const float*)d_in[11];
  const float* lb = (const float*)d_in[12];
  float* outp = (float*)d_out;

  const size_t NPTS_T = (size_t)BATCH * NPTS;                 // 16384 points
  char* w = (char*)d_ws;
  _Float16* h16  = (_Float16*)w;  w += NPTS_T * DIM * sizeof(_Float16);  // 2 MB
  _Float16* q16  = (_Float16*)w;  w += NPTS_T * DIM * sizeof(_Float16);  // 2 MB
  _Float16* k16  = (_Float16*)w;  w += NPTS_T * DIM * sizeof(_Float16);  // 2 MB
  float*    v32  = (float*)w;     w += NPTS_T * DIM * sizeof(float);     // 4 MB
  float*    nrm  = (float*)w;     w += NPTS_T * sizeof(float);           // 64 KB
  float*    o32  = (float*)w;     w += NPTS_T * DIM * sizeof(float);     // 4 MB

  k1_mlp_pool<<<NPTS_T / 16, 32, 0, stream>>>(x, w1, b1, w2, b2, h16);
  k2_qkv<<<NPTS_T / 16, 32, 0, stream>>>(h16, wq, bq, wk, bk, wv, bv, q16, k16, v32, nrm);
  k3_attn_knn<<<BATCH * (NPTS / 16), 32, 0, stream>>>(q16, k16, h16, nrm, v32, o32);
  k4_pool_linear<<<BATCH, 256, 0, stream>>>(o32, lw, lb, outp);
}
